// EASYRNN_64527588655573
// MI455X (gfx1250) — compile-verified, matrix operand order checked by reading, not run on hardware
//
#include <hip/hip_runtime.h>
#include <hip/hip_bf16.h>

// ---------------------------------------------------------------------------
// EASYRNN on gfx1250: 10-layer tanh RNN (B=4096, T=28, IN=28, H=128) + FC(10)
// Strategy:
//   * per layer: (1) input-projection GEMM over all B*T rows with
//     v_wmma_f32_16x16x32_f16 (f16 A/B, f32 acc, biases folded in, f32 out)
//     (2) recurrent scan kernel: each workgroup owns 32 batch rows, keeps
//     Whh (f16) + h-tile in LDS, loops t internally: h = tanh(xp_t + h*Whh^T)
//   * activations between layers stored as f16 (halves HBM traffic),
//     xp kept f32 for accuracy.
//   * branch-free tanh via hardware v_exp_f32/v_rcp_f32 (no libm EXEC churn).
// ---------------------------------------------------------------------------

typedef _Float16 v8h  __attribute__((ext_vector_type(8)));
typedef _Float16 v16h __attribute__((ext_vector_type(16)));
typedef float    v8f  __attribute__((ext_vector_type(8)));

union FragU { v16h v; v8h h[2]; };

// Load a 16x32 f16 WMMA A/B fragment from row-major memory. Caller passes the
// per-lane pointer: row base + khalf*8, where row = (A row) or (B output col
// n when loading from row-major W[n,k]).
__device__ __forceinline__ v16h load_frag(const _Float16* p) {
  FragU u;
  u.h[0] = *(const v8h*)(p);        // K = khalf*8 .. +8
  u.h[1] = *(const v8h*)(p + 16);   // K = 16 + khalf*8 .. +8
  return u.v;
}

// Branch-free tanh: 1 - 2/(exp2(2*log2e*x)+1). Saturates correctly at +/-inf,
// ~1e-7 abs error -- far below the f16-matmul noise floor. Pure VALU, no
// EXEC-mask manipulation (keeps the WMMA scan loop convergent).
__device__ __forceinline__ float fast_tanh(float x) {
  float e = __builtin_amdgcn_exp2f(x * 2.8853900817779268f);  // 2*log2(e)
  return 1.0f - 2.0f * __builtin_amdgcn_rcpf(e + 1.0f);
}

// ---------------------------------------------------------------------------
// Kernel 0: convert x (f32, [M,28]) to f16 padded to [M,32] (K pad for WMMA)
// ---------------------------------------------------------------------------
__global__ __launch_bounds__(256) void pad_x_kernel(
    const float* __restrict__ x, _Float16* __restrict__ xpad, int total) {
  int i = blockIdx.x * 256 + threadIdx.x;
  if (i >= total) return;
  int r = i >> 5, k = i & 31;
  xpad[i] = (k < 28) ? (_Float16)x[r * 28 + k] : (_Float16)0.f;
}

// ---------------------------------------------------------------------------
// Input projection GEMM: xp[r,n] = sum_k A[r,k]*W[n,k] + bih[n] + bhh[n]
//   A: f16 row-major [M, KS*32], W: f32 row-major [128, KW] (KW<=KS*32)
//   M = 114688, N = 128. Block = 256 threads (8 waves), tile 128 rows/block;
//   wave w handles rows w*16..+16, all 128 cols (8 n-tiles x KS k-steps).
// ---------------------------------------------------------------------------
template <int KS, int KW>
__global__ __launch_bounds__(256) void proj_kernel(
    const _Float16* __restrict__ A, const float* __restrict__ W,
    const float* __restrict__ bih, const float* __restrict__ bhh,
    float* __restrict__ xp) {
  constexpr int KD = KS * 32;
  constexpr int LW = KD + 8;  // pad rows to dodge LDS bank conflicts
  __shared__ __align__(16) _Float16 Wl[128 * LW];
  __shared__ float bl[128];

  const int tid = threadIdx.x;
  for (int i = tid; i < 128 * KD; i += 256) {
    int r = i / KD, k = i % KD;
    Wl[r * LW + k] = (k < KW) ? (_Float16)W[r * KW + k] : (_Float16)0.f;
  }
  if (tid < 128) bl[tid] = bih[tid] + bhh[tid];
  __syncthreads();

  const int lane  = tid & 31;
  const int wave  = tid >> 5;
  const int mloc  = lane & 15;
  const int khalf = lane >> 4;
  const size_t row0 = (size_t)blockIdx.x * 128 + wave * 16;

  // A fragments for this wave's 16 rows (reused across all 8 n-tiles)
  v16h af[KS];
#pragma unroll
  for (int ks = 0; ks < KS; ks++)
    af[ks] = load_frag(A + (row0 + mloc) * KD + ks * 32 + khalf * 8);

#pragma unroll
  for (int nt = 0; nt < 8; nt++) {
    const int n = nt * 16 + mloc;   // C-layout: all 8 acc elems share col n
    const float bv = bl[n];
    v8f acc = {bv, bv, bv, bv, bv, bv, bv, bv};
#pragma unroll
    for (int ks = 0; ks < KS; ks++) {
      v16h bf = load_frag(&Wl[n * LW + ks * 32 + khalf * 8]);
      acc = __builtin_amdgcn_wmma_f32_16x16x32_f16(
          false, af[ks], false, bf, (short)0, acc, false, false);
    }
    const size_t rb = row0 + 8 * khalf;  // C-layout: VGPR r -> row rb + r
#pragma unroll
    for (int r = 0; r < 8; r++) xp[(rb + r) * 128 + n] = acc[r];
  }
}

// ---------------------------------------------------------------------------
// Recurrent scan: per workgroup, 32 batch rows; loop t = 0..27 internally.
//   h = tanh(xp[:,t,:] + h @ Whh^T), h kept in LDS (f16), Whh staged in LDS.
//   8 waves: mi = wave&1 (16-row slab), nq = wave>>1 (32-col slab, 2 n-tiles).
//   Each timestep: 8 WMMAs/wave, f32 accumulate, tanh, store f16 to LDS+HBM.
// ---------------------------------------------------------------------------
__global__ __launch_bounds__(256) void recur_kernel(
    const float* __restrict__ xp,      // [B*28, 128] f32 (row = b*28 + t)
    const float* __restrict__ Whh,     // [128, 128] f32
    _Float16* __restrict__ out) {      // [B*28, 128] f16
  constexpr int LW = 136;
  __shared__ __align__(16) _Float16 Wl[128 * LW];
  __shared__ __align__(16) _Float16 hl[32 * LW];

  const int tid = threadIdx.x;
  for (int i = tid; i < 128 * 128; i += 256) {
    int r = i >> 7, k = i & 127;
    Wl[r * LW + k] = (_Float16)Whh[i];
  }
  for (int i = tid; i < 32 * LW; i += 256) hl[i] = (_Float16)0.f;
  __syncthreads();

  const int lane  = tid & 31;
  const int wave  = tid >> 5;
  const int mi    = wave & 1;        // row slab (16 rows)
  const int nq    = wave >> 1;       // col slab (32 cols)
  const int mloc  = lane & 15;
  const int khalf = lane >> 4;
  const int mrow  = mi * 16;
  const size_t bb = (size_t)blockIdx.x * 32;   // batch base

  for (int t = 0; t < 28; t++) {
    // A fragments: this slab's 16 h-rows, K = 0..127 from LDS h-tile
    v16h af[4];
#pragma unroll
    for (int ks = 0; ks < 4; ks++)
      af[ks] = load_frag(&hl[(mrow + mloc) * LW + ks * 32 + khalf * 8]);

    v8f acc[2];
#pragma unroll
    for (int nt = 0; nt < 2; nt++) {
      const int n = nq * 32 + nt * 16 + mloc;
      const size_t b0 = bb + mrow + 8 * khalf;   // batch row of acc elem r=0
      const size_t rbase = (b0 * 28 + t) * 128 + n;
#pragma unroll
      for (int r = 0; r < 8; r++)
        acc[nt][r] = xp[rbase + (size_t)r * 28 * 128];
#pragma unroll
      for (int ks = 0; ks < 4; ks++) {
        v16h bf = load_frag(&Wl[n * LW + ks * 32 + khalf * 8]);
        acc[nt] = __builtin_amdgcn_wmma_f32_16x16x32_f16(
            false, af[ks], false, bf, (short)0, acc[nt], false, false);
      }
#pragma unroll
      for (int r = 0; r < 8; r++) acc[nt][r] = fast_tanh(acc[nt][r]);
    }

    __syncthreads();  // all waves done reading hl for this t
#pragma unroll
    for (int nt = 0; nt < 2; nt++) {
      const int n = nq * 32 + nt * 16 + mloc;
#pragma unroll
      for (int r = 0; r < 8; r++) {
        _Float16 hv = (_Float16)acc[nt][r];
        hl[(mrow + 8 * khalf + r) * LW + n] = hv;
        out[((bb + mrow + 8 * khalf + r) * 28 + t) * 128 + n] = hv;
      }
    }
    __syncthreads();  // h tile fully updated before next timestep reads
  }
}

// ---------------------------------------------------------------------------
// Final FC on last timestep: out[b,c] = act[(b*28+27), :] . fcW[c,:] + fcb[c]
// ---------------------------------------------------------------------------
__global__ __launch_bounds__(256) void fc_kernel(
    const _Float16* __restrict__ act, const float* __restrict__ fcW,
    const float* __restrict__ fcb, float* __restrict__ out) {
  int i = blockIdx.x * 256 + threadIdx.x;
  if (i >= 4096 * 10) return;
  int b = i / 10, c = i % 10;
  const _Float16* a = act + ((size_t)b * 28 + 27) * 128;
  const float* w = fcW + c * 128;
  float s = fcb[c];
#pragma unroll 8
  for (int h = 0; h < 128; h++) s += (float)a[h] * w[h];
  out[i] = s;
}

// ---------------------------------------------------------------------------
extern "C" void kernel_launch(void* const* d_in, const int* in_sizes, int n_in,
                              void* d_out, int out_size, void* d_ws,
                              size_t ws_size, hipStream_t stream) {
  (void)in_sizes; (void)n_in; (void)out_size; (void)ws_size;
  const float* x    = (const float*)d_in[0];   // [4096,28,28]
  const float* Wih0 = (const float*)d_in[1];   // [128,28]
  const float* Wih  = (const float*)d_in[2];   // [9,128,128]
  const float* Whh  = (const float*)d_in[3];   // [10,128,128]
  const float* bih  = (const float*)d_in[4];   // [10,128]
  const float* bhh  = (const float*)d_in[5];   // [10,128]
  const float* fcW  = (const float*)d_in[6];   // [10,128]
  const float* fcb  = (const float*)d_in[7];   // [10]
  float* out = (float*)d_out;                  // [4096,10]

  const size_t M = (size_t)4096 * 28;          // 114688 rows
  char* ws = (char*)d_ws;
  _Float16* xpad = (_Float16*)ws; ws += M * 32 * sizeof(_Float16);
  _Float16* actA = (_Float16*)ws; ws += M * 128 * sizeof(_Float16);
  _Float16* actB = (_Float16*)ws; ws += M * 128 * sizeof(_Float16);
  float*    xp   = (float*)ws;                 // M*128 f32

  // x -> f16 padded [M,32]
  pad_x_kernel<<<(int)((M * 32 + 255) / 256), 256, 0, stream>>>(
      x, xpad, (int)(M * 32));

  // layer 0: K=28 (padded to 32)
  proj_kernel<1, 28><<<(int)(M / 128), 256, 0, stream>>>(
      xpad, Wih0, bih, bhh, xp);
  recur_kernel<<<4096 / 32, 256, 0, stream>>>(xp, Whh, actA);

  _Float16* cur = actA;
  _Float16* nxt = actB;
  for (int l = 1; l < 10; l++) {
    proj_kernel<4, 128><<<(int)(M / 128), 256, 0, stream>>>(
        cur, Wih + (size_t)(l - 1) * 128 * 128, bih + l * 128, bhh + l * 128,
        xp);
    recur_kernel<<<4096 / 32, 256, 0, stream>>>(
        xp, Whh + (size_t)l * 128 * 128, nxt);
    _Float16* tmp = cur; cur = nxt; nxt = tmp;
  }

  fc_kernel<<<(4096 * 10 + 255) / 256, 256, 0, stream>>>(cur, fcW, fcb, out);
}
